// EpisodicMemoryEfficient_53042846106049
// MI455X (gfx1250) — compile-verified, matrix operand order checked by reading, not run on hardware
//
#include <hip/hip_runtime.h>
#include <math.h>

typedef _Float16 h16;
typedef __attribute__((ext_vector_type(16))) _Float16 v16h;
typedef __attribute__((ext_vector_type(8)))  float    v8f;

#define D_DIM  512
#define S_LEN  8192
#define BATCH  4
#define CHUNK  64
#define NCHUNK (S_LEN / CHUNK)   // 128
#define MTOT   (BATCH * S_LEN)   // 32768

// ---------------------------------------------------------------------------
// CDNA5 async global->LDS copy (ASYNCcnt-tracked, bypasses VGPRs).
// LDS destination = low 32 bits of the generic shared-space pointer
// ---------------------------------------------------------------------------
static __device__ __forceinline__ unsigned lds_off(const void* p) {
  return (unsigned)(uintptr_t)p;
}
static __device__ __forceinline__ void async_cp_b128(unsigned dst_lds,
                                                     const void* src) {
  unsigned long long ga = (unsigned long long)(uintptr_t)src;
  asm volatile("global_load_async_to_lds_b128 %0, %1, off"
               :: "v"(dst_lds), "v"(ga)
               : "memory");
}
static __device__ __forceinline__ void wait_async0() {
  asm volatile("s_wait_asynccnt 0x0" ::: "memory");
}

// ---------------------------------------------------------------------------
// Load a 16x32 f16 fragment (A layout; B uses the mirrored layout with
// lane = N column) from an LDS tile stored row-major with `stride` halves.
// ISA 7.12.2 (16-bit A 16x32): lane&15 -> M, lane>>4 -> K half (+8);
// VGPR r<4: K = 2r + 8*half ; r>=4: K = 16 + 2(r-4) + 8*half.
// ---------------------------------------------------------------------------
static __device__ __forceinline__ v16h frag_ld(const h16* base, int stride,
                                               int row0, int col0, int lane) {
  const int hf = (lane >> 4) & 1;
  const int i  = lane & 15;
  union { v16h v; unsigned u[8]; } f;
#pragma unroll
  for (int r = 0; r < 8; ++r) {
    const int kb = ((r < 4) ? 0 : 16) + 8 * hf + 2 * (r & 3);
    f.u[r] = *(const unsigned*)(base + (size_t)(row0 + i) * stride + col0 + kb);
  }
  return f.v;
}

// B fragment (32(K) x 16(N)) built from a ROW-MAJOR tile [32 K][128 N]:
// lane covers column n = dcol + (lane&15); pack the K-pair (j, j+1) from two
// rows. Avoids any LDS-side transpose staging.
static __device__ __forceinline__ v16h frag_ld_pairK(const h16* base, int dcol,
                                                     int lane) {
  const int hf = (lane >> 4) & 1;
  const int n  = dcol + (lane & 15);
  union { v16h v; unsigned u[8]; } f;
#pragma unroll
  for (int r = 0; r < 8; ++r) {
    const int j = ((r < 4) ? 0 : 16) + 8 * hf + 2 * (r & 3);
    const unsigned short lo =
        *(const unsigned short*)(base + (size_t)j * 128 + n);
    const unsigned short hi =
        *(const unsigned short*)(base + (size_t)(j + 1) * 128 + n);
    f.u[r] = (unsigned)lo | ((unsigned)hi << 16);
  }
  return f.v;
}

static __device__ __forceinline__ v8f wmma_f16(v16h a, v16h b, v8f c) {
  return __builtin_amdgcn_wmma_f32_16x16x32_f16(false, a, false, b,
                                                (short)0, c, false, false);
}

// ---------------------------------------------------------------------------
// Kernel 1: q/k/v = x @ W^T, output f16. x:[32768,512] f32, W:[512,512] f32.
// Block: 128 threads (4 waves), 64x64 output tile. grid = (512, 8, 3).
// ---------------------------------------------------------------------------
__global__ void qkv_gemm_kernel(const float* __restrict__ x,
                                const float* __restrict__ Wq,
                                const float* __restrict__ Wk,
                                const float* __restrict__ Wv,
                                h16* __restrict__ q,
                                h16* __restrict__ k,
                                h16* __restrict__ v) {
  const int z = blockIdx.z;
  const float* __restrict__ W = (z == 0) ? Wq : (z == 1) ? Wk : Wv;
  h16* __restrict__ o         = (z == 0) ? q  : (z == 1) ? k  : v;

  const int m0 = blockIdx.x * 64;
  const int n0 = blockIdx.y * 64;
  const int tid = threadIdx.x, lane = tid & 31, wv = tid >> 5;

  __shared__ __align__(16) h16 sA[64 * 32];
  __shared__ __align__(16) h16 sB[64 * 32];

  v8f acc[4] = {};

  for (int k0 = 0; k0 < D_DIM; k0 += 32) {
#pragma unroll
    for (int it = 0; it < 4; ++it) {
      const int idx = tid + 128 * it;
      const int r = idx >> 3, c4 = (idx & 7) * 4;
      const float4 f = *(const float4*)(x + (size_t)(m0 + r) * D_DIM + k0 + c4);
      sA[r * 32 + c4 + 0] = (h16)f.x;
      sA[r * 32 + c4 + 1] = (h16)f.y;
      sA[r * 32 + c4 + 2] = (h16)f.z;
      sA[r * 32 + c4 + 3] = (h16)f.w;
    }
#pragma unroll
    for (int it = 0; it < 4; ++it) {
      const int idx = tid + 128 * it;
      const int r = idx >> 3, c4 = (idx & 7) * 4;
      const float4 f = *(const float4*)(W + (size_t)(n0 + r) * D_DIM + k0 + c4);
      sB[r * 32 + c4 + 0] = (h16)f.x;
      sB[r * 32 + c4 + 1] = (h16)f.y;
      sB[r * 32 + c4 + 2] = (h16)f.z;
      sB[r * 32 + c4 + 3] = (h16)f.w;
    }
    if (k0 + 32 < D_DIM) {  // prefetch next K-tile (global_prefetch_b8)
      __builtin_prefetch(x + (size_t)(m0 + (tid >> 1)) * D_DIM + k0 + 32, 0, 0);
      __builtin_prefetch(W + (size_t)(n0 + (tid >> 1)) * D_DIM + k0 + 32, 0, 0);
    }
    __syncthreads();

    const v16h a = frag_ld(sA, 32, 16 * wv, 0, lane);
#pragma unroll
    for (int t = 0; t < 4; ++t) {
      const v16h bm = frag_ld(sB, 32, 16 * t, 0, lane);
      acc[t] = wmma_f16(a, bm, acc[t]);
    }
    __syncthreads();
  }

  const int hf = (lane >> 4) & 1, nl = lane & 15;
#pragma unroll
  for (int t = 0; t < 4; ++t)
#pragma unroll
    for (int r = 0; r < 8; ++r)
      o[(size_t)(m0 + 16 * wv + r + 8 * hf) * D_DIM + n0 + 16 * t + nl] =
          (h16)acc[t][r];
}

// ---------------------------------------------------------------------------
// Kernel 2: chunked sliding-window attention, one block per (batch, chunk).
// Block: 128 threads (4 waves). C=64 queries, 2C=128 context keys, D=512.
// All f16 tile staging (Q, K, V) uses double-buffered
// GLOBAL_LOAD_ASYNC_TO_LDS_B128 with one s_wait_asynccnt + one barrier per
// pipeline step.
//
// LDS pool (48 KB):
//   region A (0..32 KB): phase-1 buffers Q[2][64*32]+K[2][128*32] (24 KB)
//                        -> scores f32 [64*128] (32 KB)
//                        -> phase-2 V buffers [2][32*128] f16 (16 KB)
//   sP (32..48 KB): probabilities f16 [64*128]
// ---------------------------------------------------------------------------
__global__ void attn_kernel(const h16* __restrict__ q,
                            const h16* __restrict__ kk,
                            const h16* __restrict__ vv,
                            float* __restrict__ out) {
  const int blk = blockIdx.x;
  const int b = blk / NCHUNK, ch = blk % NCHUNK;
  const int tid = threadIdx.x, lane = tid & 31, wv = tid >> 5;

  __shared__ __align__(16) char smem[48 * 1024];
  float* sS = (float*)smem;            // 64*128 f32 (after phase 1)
  h16*   sP = (h16*)(smem + 32768);    // 64*128 f16

  const size_t qbase = ((size_t)b * S_LEN + (size_t)ch * CHUNK) * D_DIM;
  const int chm1 = (ch > 0) ? ch - 1 : ch;  // dummy (fully masked) for ch==0

  auto krow = [&](int j) -> size_t {
    const int cc = (j < CHUNK) ? chm1 : ch;
    const int jj = (j < CHUNK) ? j : j - CHUNK;
    return ((size_t)b * S_LEN + (size_t)cc * CHUNK + jj) * D_DIM;
  };

  auto qbuf = [&](int buf) -> h16* { return (h16*)(smem + buf * 4096); };
  auto kbuf = [&](int buf) -> h16* { return (h16*)(smem + 8192 + buf * 8192); };
  auto vbuf = [&](int buf) -> h16* { return (h16*)(smem + buf * 8192); };

  // ---------------- Phase 1: scores = Q @ Kctx^T ----------------
  auto stage_qk = [&](int buf, int k0) {
    h16* dQ = qbuf(buf);
    h16* dK = kbuf(buf);
#pragma unroll
    for (int it = 0; it < 2; ++it) {          // Q tile: 64x32 halves
      const int idx = tid + 128 * it;
      const int r = idx >> 2, seg = (idx & 3) * 8;
      async_cp_b128(lds_off(dQ + r * 32 + seg),
                    q + qbase + (size_t)r * D_DIM + k0 + seg);
    }
#pragma unroll
    for (int it = 0; it < 4; ++it) {          // K tile: 128x32 halves
      const int idx = tid + 128 * it;
      const int r = idx >> 2, seg = (idx & 3) * 8;
      async_cp_b128(lds_off(dK + r * 32 + seg),
                    kk + krow(r) + k0 + seg);
    }
  };

  v8f acc[8] = {};
  stage_qk(0, 0);
  for (int s = 0; s < 16; ++s) {
    wait_async0();          // this wave's tile-s copies landed
    __syncthreads();        // everyone's copies landed; prev readers done
    if (s < 15) stage_qk((s + 1) & 1, 32 * (s + 1));

    const h16* cQ = qbuf(s & 1);
    const h16* cK = kbuf(s & 1);
    const v16h a = frag_ld(cQ, 32, 16 * wv, 0, lane);
#pragma unroll
    for (int t = 0; t < 8; ++t) {
      const v16h bm = frag_ld(cK, 32, 16 * t, 0, lane);
      acc[t] = wmma_f16(a, bm, acc[t]);
    }
  }
  __syncthreads();          // staging buffers dead; safe to overwrite with sS

  // dump scaled scores
  {
    const float scale = 0.044194173824159216f;  // 1/sqrt(512)
    const int hf = (lane >> 4) & 1, nl = lane & 15;
#pragma unroll
    for (int t = 0; t < 8; ++t)
#pragma unroll
      for (int r = 0; r < 8; ++r)
        sS[(size_t)(16 * wv + r + 8 * hf) * 128 + 16 * t + nl] =
            acc[t][r] * scale;
  }
  __syncthreads();

  // ---------------- masked softmax ----------------
  if (tid < 64) {
    const int qi = tid;
    float mx = -3.0e38f;
    int cnt = 0;
    for (int j = 0; j < 128; ++j) {
      const bool valid = (j < CHUNK + qi) && (ch > 0 || j >= CHUNK);
      if (valid) {
        const float s = sS[qi * 128 + j];
        mx = (s > mx) ? s : mx;
        ++cnt;
      }
    }
    float sum = 0.f;
    for (int j = 0; j < 128; ++j) {
      const bool valid = (j < CHUNK + qi) && (ch > 0 || j >= CHUNK);
      const float e = valid ? __expf(sS[qi * 128 + j] - mx) : 0.f;
      sS[qi * 128 + j] = e;
      sum += e;
    }
    const float inv = (cnt > 0) ? (1.f / sum) : 0.f;  // empty row -> zeros
    for (int j = 0; j < 128; ++j)
      sP[qi * 128 + j] = (h16)(sS[qi * 128 + j] * inv);
  }
  __syncthreads();          // sS dead after this point

  // ---------------- Phase 2: O = P @ Vctx ----------------
  // Flattened 16-step pipeline: step -> (d0 = (step>>2)*128, j0 = (step&3)*32).
  // V tile staged ROW-MAJOR [32 j][128 d] by async copy; B fragments pack
  // K-pairs from two rows (no transpose staging).
  auto stage_v = [&](int buf, int step) {
    const int d0 = (step >> 2) * 128;
    const int j0 = (step & 3) * 32;
    h16* dV = vbuf(buf);
#pragma unroll
    for (int it = 0; it < 4; ++it) {          // 32x128 halves = 512 x b128
      const int idx = tid + 128 * it;
      const int r = idx >> 4, seg = (idx & 15) * 8;
      async_cp_b128(lds_off(dV + r * 128 + seg),
                    vv + krow(j0 + r) + d0 + seg);
    }
  };

  v8f oacc[8] = {};
  stage_v(0, 0);
  for (int step = 0; step < 16; ++step) {
    const int d0 = (step >> 2) * 128;
    const int j0 = (step & 3) * 32;
    const int js = step & 3;

    wait_async0();
    __syncthreads();
    if (step < 15) stage_v((step + 1) & 1, step + 1);

    if (js == 0) {
#pragma unroll
      for (int t = 0; t < 8; ++t) oacc[t] = (v8f){};
    }

    const h16* cV = vbuf(step & 1);
    const v16h a = frag_ld(sP, 128, 16 * wv, j0, lane);
#pragma unroll
    for (int t = 0; t < 8; ++t) {
      const v16h bm = frag_ld_pairK(cV, 16 * t, lane);
      oacc[t] = wmma_f16(a, bm, oacc[t]);
    }

    if (js == 3) {  // d-block complete: write fp32 output
      const int hf = (lane >> 4) & 1, nl = lane & 15;
#pragma unroll
      for (int t = 0; t < 8; ++t)
#pragma unroll
        for (int r = 0; r < 8; ++r)
          out[qbase + (size_t)(16 * wv + r + 8 * hf) * D_DIM + d0 + 16 * t +
              nl] = oacc[t][r];
    }
  }
}

// ---------------------------------------------------------------------------
extern "C" void kernel_launch(void* const* d_in, const int* in_sizes, int n_in,
                              void* d_out, int out_size, void* d_ws,
                              size_t ws_size, hipStream_t stream) {
  const float* x  = (const float*)d_in[0];  // states [4,8192,256,2]
  const float* Wq = (const float*)d_in[1];
  const float* Wk = (const float*)d_in[2];
  const float* Wv = (const float*)d_in[3];

  const size_t elems = (size_t)MTOT * D_DIM;
  h16* q = (h16*)d_ws;
  h16* k = q + elems;
  h16* v = k + elems;  // ws use: 96 MB

  dim3 g1(MTOT / 64, D_DIM / 64, 3);
  qkv_gemm_kernel<<<g1, 128, 0, stream>>>(x, Wq, Wk, Wv, q, k, v);

  attn_kernel<<<dim3(BATCH * NCHUNK), 128, 0, stream>>>(q, k, v,
                                                        (float*)d_out);
}
